// ASTNN_30648886624757
// MI455X (gfx1250) — matrix operand order repaired and checked by reading, not compiled, not run on hardware
//
#include <hip/hip_runtime.h>
#include <math.h>

// ---------------------------------------------------------------------------
// Types
// ---------------------------------------------------------------------------
typedef __bf16 bf16;
typedef __attribute__((ext_vector_type(16))) __bf16 v16bf;
typedef __attribute__((ext_vector_type(8)))  __bf16 v8bf;
typedef __attribute__((ext_vector_type(8)))  float  v8f;

#define B_  64
#define L_  64
#define N_  32
#define E_  128
#define C_  128
#define H_  100
#define OUT_ 104
#define G3_ 300          // 3*H
#define NTG 19           // ceil(304/16) n-tiles for the 3H=300 dim (padded to 304)

// CDNA5 async global->LDS loads (ASYNCcnt-tracked). Guarded so the host pass
// and any toolchain lacking the builtin fall back to the synchronous path.
#if defined(__gfx1250__) && __has_builtin(__builtin_amdgcn_global_load_async_to_lds_b128) && __has_builtin(__builtin_amdgcn_s_wait_asynccnt)
#define HAVE_ASYNC_LDS 1
#else
#define HAVE_ASYNC_LDS 0
#endif

__device__ __forceinline__ bf16 f2bf(float x) {
    unsigned u = __float_as_uint(x);
    u += 0x7fffu + ((u >> 16) & 1u);           // round-to-nearest-even
    unsigned short h = (unsigned short)(u >> 16);
    return __builtin_bit_cast(bf16, h);
}

#if HAVE_ASYNC_LDS
// Builtin signature (from hipcc diagnostic): param0 = v4i addrspace(1)*,
// param1 = v4i addrspace(3)*, then two immediate ints (offset, cpol).
typedef int v4i_vs __attribute__((vector_size(16)));
__device__ __forceinline__ void async_copy16(const void* gptr, void* lptr) {
    __builtin_amdgcn_global_load_async_to_lds_b128(
        (__attribute__((address_space(1))) v4i_vs*)(unsigned long long)(uintptr_t)gptr,
        (__attribute__((address_space(3))) v4i_vs*)(unsigned int)(uintptr_t)lptr,
        0, 0);
}
#endif

// Load a 16x16x32 bf16 A-fragment for tile-row m (= lane%16) from a row-major
// bf16 matrix with row stride 128, K window starting at k0.
__device__ __forceinline__ v16bf load_a_frag(const bf16* base, int mrow, int k0, int lane) {
    int kb = k0 + ((lane < 16) ? 0 : 16);
    const bf16* ap = base + mrow * 128 + kb;
    v8bf lo = *(const v8bf*)ap;
    v8bf hi = *(const v8bf*)(ap + 8);
    v16bf a;
#pragma unroll
    for (int i = 0; i < 8; ++i) { a[i] = lo[i]; a[8 + i] = hi[i]; }
    return a;
}

// ---------------------------------------------------------------------------
// Kernel 0: weight prep (transpose W_lin to bf16; zero-pad w_hh to [304][128] bf16)
// ---------------------------------------------------------------------------
__global__ __launch_bounds__(256) void k_prep(
    const float* __restrict__ W_lin,
    const float* __restrict__ whh_f, const float* __restrict__ whh_b,
    bf16* __restrict__ Wt, bf16* __restrict__ whhp_f, bf16* __restrict__ whhp_b)
{
    int idx = blockIdx.x * blockDim.x + threadIdx.x;
    int stride = gridDim.x * blockDim.x;
    for (int i = idx; i < 128 * 128; i += stride) {
        int n = i >> 7, k = i & 127;
        Wt[i] = f2bf(W_lin[k * 128 + n]);      // Wt[n][k] = W_lin[k][n]
    }
    for (int i = idx; i < 304 * 128; i += stride) {
        int n = i >> 7, k = i & 127;
        bool ok = (n < G3_) && (k < H_);
        whhp_f[i] = f2bf(ok ? whh_f[n * H_ + k] : 0.f);
        whhp_b[i] = f2bf(ok ? whh_b[n * H_ + k] : 0.f);
    }
}

// ---------------------------------------------------------------------------
// Kernel 1: per-tree  h = emb[ids] @ W_lin + b  -> child-sum (desc. index) -> max
// One block per tree (32 nodes x 128 ch). 256 threads = 8 waves.
// Embedding rows are gathered memory->LDS with async loads (no VGPR round-trip).
// ---------------------------------------------------------------------------
__global__ __launch_bounds__(256) void k_tree(
    const int* __restrict__ node_ids, const int* __restrict__ parent,
    const float* __restrict__ emb, const bf16* __restrict__ Wt,
    const float* __restrict__ b_lin, float* __restrict__ stmt)
{
    __shared__ __align__(16) float stageF[N_ * 128]; // 16 KB f32 gather staging
    __shared__ __align__(32) bf16 As[N_ * 128];      // 8 KB
    __shared__ float hs[N_ * 128];                   // 16 KB
    __shared__ int par[N_];

    const int tree = blockIdx.x;
    const int tid = threadIdx.x, lane = tid & 31, wave = tid >> 5;
    const int base = tree * N_;

    if (tid < N_) par[tid] = parent[base + tid] - base;

#if HAVE_ASYNC_LDS
    // 32 rows x 512 B = 1024 async 16-B chunks, issued by all 256 threads.
    for (int c = tid; c < N_ * 32; c += 256) {
        int node = c >> 5;                            // 32 chunks per row
        int id = node_ids[base + node];
        const float* g = emb + (size_t)id * 128 + (c & 31) * 4;
        async_copy16(g, &stageF[c * 4]);
    }
    __builtin_amdgcn_s_wait_asynccnt(0);
    __syncthreads();
    for (int i = tid; i < N_ * 128; i += 256)
        As[i] = f2bf(stageF[i]);
#else
    for (int i = tid; i < N_ * 128; i += 256) {
        int node = i >> 7, c = i & 127;
        int id = node_ids[base + node];
        As[i] = f2bf(emb[(size_t)id * 128 + c]);
    }
#endif
    __syncthreads();

    // GEMM: 2 m-tiles x 8 n-tiles x (K=128 -> 4 steps of 32)
    const int mt = wave & 1;
#pragma unroll
    for (int q = 0; q < 2; ++q) {
        const int nt = (wave >> 1) + q * 4;
        const int n = nt * 16 + (lane & 15);
        v8f acc = {};
#pragma unroll
        for (int kt = 0; kt < 4; ++kt) {
            v16bf a = load_a_frag(As, mt * 16 + (lane & 15), kt * 32, lane);
            int kk = kt * 32 + ((lane < 16) ? 0 : 16);
            v16bf b = *(const v16bf*)(Wt + n * 128 + kk);
            acc = __builtin_amdgcn_wmma_f32_16x16x32_bf16(
                false, a, false, b, (short)0, acc, false, false);
        }
        const float bias = b_lin[n];
#pragma unroll
        for (int r = 0; r < 8; ++r) {
            int m = mt * 16 + r + ((lane < 16) ? 0 : 8);
            hs[m * 128 + n] = acc[r] + bias;
        }
    }
    __syncthreads();

    // Child-sum: parent[i] < i, so a single descending pass == bottom-up by level.
    if (tid < 128) {
        const int c = tid;
        for (int i = N_ - 1; i >= 1; --i)
            hs[par[i] * 128 + c] += hs[i * 128 + c];
        float mx = hs[c];
#pragma unroll 4
        for (int i = 1; i < N_; ++i) mx = fmaxf(mx, hs[i * 128 + c]);
        stmt[tree * 128 + c] = mx;
    }
}

// ---------------------------------------------------------------------------
// Kernel 2: gi = stmt @ w_ih^T + b_ih for both directions.
// grid (64 row-blocks, 2 dirs); 256 threads = 8 waves; M=64/block, N=300(->304), K=128.
// ---------------------------------------------------------------------------
__global__ __launch_bounds__(256) void k_gi(
    const float* __restrict__ stmt,
    const float* __restrict__ wih_f, const float* __restrict__ bih_f,
    const float* __restrict__ wih_b, const float* __restrict__ bih_b,
    float* __restrict__ gi_f, float* __restrict__ gi_b)
{
    __shared__ __align__(16) float stageF[64 * 128]; // 32 KB f32 staging
    __shared__ __align__(32) bf16 As[64 * 128];      // 16 KB

    const int dir = blockIdx.y;
    const float* wih = dir ? wih_b : wih_f;
    const float* bih = dir ? bih_b : bih_f;
    float* gi = dir ? gi_b : gi_f;

    const int tid = threadIdx.x, lane = tid & 31, wave = tid >> 5;
    const int row0 = blockIdx.x * 64;

#if HAVE_ASYNC_LDS
    // Tile is fully contiguous: 64 rows x 512 B = 2048 async 16-B chunks.
    for (int c = tid; c < 64 * 32; c += 256) {
        const float* g = stmt + (size_t)row0 * 128 + c * 4;
        async_copy16(g, &stageF[c * 4]);
    }
    __builtin_amdgcn_s_wait_asynccnt(0);
    __syncthreads();
    for (int i = tid; i < 64 * 128; i += 256)
        As[i] = f2bf(stageF[i]);
#else
    for (int i = tid; i < 64 * 128; i += 256)
        As[i] = f2bf(stmt[(size_t)(row0 + (i >> 7)) * 128 + (i & 127)]);
#endif
    __syncthreads();

    for (int nt = wave; nt < NTG; nt += 8) {        // wave-uniform loop
        const int n = nt * 16 + (lane & 15);
        const int nr = (n < G3_) ? n : (G3_ - 1);   // clamp OOB row loads
        v16bf Bf[4];
#pragma unroll
        for (int kt = 0; kt < 4; ++kt) {
            int kk = kt * 32 + ((lane < 16) ? 0 : 16);
            const float* wr = wih + nr * 128 + kk;  // B[k][n] = w_ih[n][k] (contig.)
            v16bf b;
#pragma unroll
            for (int i = 0; i < 16; ++i) b[i] = f2bf(wr[i]);
            Bf[kt] = b;
        }
        const float bias = (n < G3_) ? bih[n] : 0.f;
#pragma unroll
        for (int mt = 0; mt < 4; ++mt) {
            v8f acc = {};
#pragma unroll
            for (int kt = 0; kt < 4; ++kt) {
                v16bf a = load_a_frag(As, mt * 16 + (lane & 15), kt * 32, lane);
                acc = __builtin_amdgcn_wmma_f32_16x16x32_bf16(
                    false, a, false, Bf[kt], (short)0, acc, false, false);
            }
            if (n < G3_) {
#pragma unroll
                for (int r = 0; r < 8; ++r) {
                    int t = row0 + mt * 16 + r + ((lane < 16) ? 0 : 8);
                    gi[(size_t)t * G3_ + n] = acc[r] + bias;
                }
            }
        }
    }
}

// ---------------------------------------------------------------------------
// Kernel 3: recurrent GRU, one block per direction, 512 threads = 16 waves.
// w_hh B-fragments preloaded once into registers; per step: h->bf16, WMMA gh,
// gate math + running max-pool in LDS. Next step's gi rows are prefetched
// (global_prefetch_b8) while the current step's WMMAs run.
// ---------------------------------------------------------------------------
#define SM_HA   0                      // bf16 [64][128]  = 16384 B
#define SM_H    16384                  // f32  [64][100]  = 25600 B
#define SM_GH   (16384 + 25600)        // f32  [64][304]  = 77824 B
#define SM_POOL (16384 + 25600 + 77824)// f32  [64][100]  = 25600 B
#define SM_TOTAL (16384 + 25600 + 77824 + 25600)

__global__ __launch_bounds__(512) void k_gru(
    const float* __restrict__ gi_f, const float* __restrict__ gi_b,
    const bf16* __restrict__ whhp_f, const bf16* __restrict__ whhp_b,
    const float* __restrict__ bhh_f, const float* __restrict__ bhh_b,
    float* __restrict__ pooled)
{
    extern __shared__ __align__(32) char smem[];
    bf16*  hA    = (bf16*)(smem + SM_HA);
    float* hS    = (float*)(smem + SM_H);
    float* ghS   = (float*)(smem + SM_GH);
    float* poolS = (float*)(smem + SM_POOL);

    const int dir = blockIdx.x;
    const float* gi  = dir ? gi_b  : gi_f;
    const bf16*  whh = dir ? whhp_b : whhp_f;
    const float* bhh = dir ? bhh_b : bhh_f;

    const int tid = threadIdx.x, lane = tid & 31, wave = tid >> 5;

    for (int i = tid; i < B_ * H_; i += 512) { hS[i] = 0.f; poolS[i] = -3.402823466e38f; }

    // Preload w_hh B-fragments (reused for all 64 steps). Wave w owns n-tile w,
    // waves 0..2 additionally own tiles 16..18.
    const int nOwn = (wave < NTG - 16) ? 2 : 1;
    v16bf Bf[2][4];
#pragma unroll
    for (int t = 0; t < 2; ++t) {
        if (t >= nOwn) continue;                    // wave-uniform
        int nt = (t == 0) ? wave : (16 + wave);
        int n = nt * 16 + (lane & 15);
#pragma unroll
        for (int kt = 0; kt < 4; ++kt) {
            int kk = kt * 32 + ((lane < 16) ? 0 : 16);
            Bf[t][kt] = *(const v16bf*)(whh + n * 128 + kk);
        }
    }
    __syncthreads();

    for (int s = 0; s < L_; ++s) {
        const int l = dir ? (L_ - 1 - s) : s;

        // Prefetch next step's gi rows into cache while this step computes.
        if (s + 1 < L_) {
            const int ln = dir ? (L_ - 2 - s) : (s + 1);
            if (tid < B_)
                __builtin_prefetch(gi + ((size_t)(tid * L_ + ln)) * G3_, 0, 1);
        }

        // h (f32, K=100) -> bf16 A-matrix [64][128], zero-padded K
        for (int i = tid; i < B_ * 128; i += 512) {
            int m = i >> 7, k = i & 127;
            hA[i] = f2bf((k < H_) ? hS[m * H_ + k] : 0.f);
        }
        __syncthreads();

        // gh = h @ w_hh^T + b_hh  via WMMA (4 m-tiles x owned n-tiles x 4 k-steps)
#pragma unroll
        for (int t = 0; t < 2; ++t) {
            if (t >= nOwn) continue;                // wave-uniform
            int nt = (t == 0) ? wave : (16 + wave);
            int n = nt * 16 + (lane & 15);
            float bias = (n < G3_) ? bhh[n] : 0.f;
#pragma unroll
            for (int mt = 0; mt < 4; ++mt) {
                v8f acc = {};
#pragma unroll
                for (int kt = 0; kt < 4; ++kt) {
                    v16bf a = load_a_frag(hA, mt * 16 + (lane & 15), kt * 32, lane);
                    acc = __builtin_amdgcn_wmma_f32_16x16x32_bf16(
                        false, a, false, Bf[t][kt], (short)0, acc, false, false);
                }
                if (n < G3_) {
#pragma unroll
                    for (int r = 0; r < 8; ++r) {
                        int m = mt * 16 + r + ((lane < 16) ? 0 : 8);
                        ghS[m * 304 + n] = acc[r] + bias;
                    }
                }
            }
        }
        __syncthreads();

        // Gates + state update + running max-pool
        for (int i = tid; i < B_ * H_; i += 512) {
            int b = i / H_, j = i % H_;
            const float* g = gi + ((size_t)(b * L_ + l)) * G3_;
            float ir = g[j], iz = g[H_ + j], inn = g[2 * H_ + j];
            float hr = ghS[b * 304 + j];
            float hz = ghS[b * 304 + H_ + j];
            float hn = ghS[b * 304 + 2 * H_ + j];
            float rg = 1.f / (1.f + __expf(-(ir + hr)));
            float zg = 1.f / (1.f + __expf(-(iz + hz)));
            float ng = tanhf(inn + rg * hn);
            float hp = hS[b * H_ + j];
            float hnew = (1.f - zg) * ng + zg * hp;
            hS[b * H_ + j] = hnew;
            poolS[i] = fmaxf(poolS[i], hnew);
        }
        __syncthreads();
    }

    for (int i = tid; i < B_ * H_; i += 512) {
        int b = i / H_, j = i % H_;
        pooled[b * (2 * H_) + dir * H_ + j] = poolS[i];
    }
}

// ---------------------------------------------------------------------------
// Kernel 4: out = pooled @ fc_w + fc_b   (64x200 @ 200x104 — tiny, VALU)
// ---------------------------------------------------------------------------
__global__ __launch_bounds__(256) void k_fc(
    const float* __restrict__ pooled, const float* __restrict__ fc_w,
    const float* __restrict__ fc_b, float* __restrict__ out)
{
    int idx = blockIdx.x * 256 + threadIdx.x;
    if (idx >= B_ * OUT_) return;
    int b = idx / OUT_, o = idx % OUT_;
    float acc = fc_b[o];
#pragma unroll 4
    for (int j = 0; j < 2 * H_; ++j)
        acc += pooled[b * (2 * H_) + j] * fc_w[j * OUT_ + o];
    out[idx] = acc;
}

// ---------------------------------------------------------------------------
// Launch
// ---------------------------------------------------------------------------
extern "C" void kernel_launch(void* const* d_in, const int* in_sizes, int n_in,
                              void* d_out, int out_size, void* d_ws, size_t ws_size,
                              hipStream_t stream) {
    const int*   node_ids = (const int*)d_in[0];
    const int*   parent   = (const int*)d_in[1];
    // d_in[2] = level (unneeded: parent<child ordering suffices), d_in[3] = tree_id
    const float* emb      = (const float*)d_in[4];
    const float* W_lin    = (const float*)d_in[5];
    const float* b_lin    = (const float*)d_in[6];
    const float* w_ih_f   = (const float*)d_in[7];
    const float* w_hh_f   = (const float*)d_in[8];
    const float* b_ih_f   = (const float*)d_in[9];
    const float* b_hh_f   = (const float*)d_in[10];
    const float* w_ih_b   = (const float*)d_in[11];
    const float* w_hh_b   = (const float*)d_in[12];
    const float* b_ih_b   = (const float*)d_in[13];
    const float* b_hh_b   = (const float*)d_in[14];
    const float* fc_w     = (const float*)d_in[15];
    const float* fc_b     = (const float*)d_in[16];
    float* out = (float*)d_out;

    char* ws = (char*)d_ws;
    const size_t OFF_WT   = 0;                       // 128*128 bf16     = 32768
    const size_t OFF_WHHF = 32768;                   // 304*128 bf16     = 77824
    const size_t OFF_WHHB = 110592;                  // 304*128 bf16     = 77824
    const size_t OFF_STMT = 188416;                  // 4096*128 f32     = 2097152
    const size_t OFF_GIF  = 2285568;                 // 4096*300 f32     = 4915200
    const size_t OFF_GIB  = 7200768;                 // 4096*300 f32     = 4915200
    const size_t OFF_POOL = 12115968;                // 64*200 f32       = 51200

    bf16*  Wt     = (bf16*)(ws + OFF_WT);
    bf16*  whhp_f = (bf16*)(ws + OFF_WHHF);
    bf16*  whhp_b = (bf16*)(ws + OFF_WHHB);
    float* stmt   = (float*)(ws + OFF_STMT);
    float* gi_f   = (float*)(ws + OFF_GIF);
    float* gi_b   = (float*)(ws + OFF_GIB);
    float* pooled = (float*)(ws + OFF_POOL);

    k_prep<<<dim3(128), dim3(256), 0, stream>>>(W_lin, w_hh_f, w_hh_b,
                                                Wt, whhp_f, whhp_b);
    k_tree<<<dim3(B_ * L_), dim3(256), 0, stream>>>(node_ids, parent, emb,
                                                    Wt, b_lin, stmt);
    k_gi<<<dim3(64, 2), dim3(256), 0, stream>>>(stmt, w_ih_f, b_ih_f,
                                                w_ih_b, b_ih_b, gi_f, gi_b);
    k_gru<<<dim3(2), dim3(512), SM_TOTAL, stream>>>(gi_f, gi_b, whhp_f, whhp_b,
                                                    b_hh_f, b_hh_b, pooled);
    k_fc<<<dim3((B_ * OUT_ + 255) / 256), dim3(256), 0, stream>>>(pooled, fc_w,
                                                                  fc_b, out);
}
